// MultiHeadedAttention_37391985279580
// MI455X (gfx1250) — matrix operand order repaired and checked by reading, not compiled, not run on hardware
//
#include <hip/hip_runtime.h>

typedef __attribute__((ext_vector_type(16))) _Float16 v16h;
typedef __attribute__((ext_vector_type(8)))  float    v8f;
typedef __attribute__((ext_vector_type(4)))  unsigned int u32x4;
typedef __attribute__((ext_vector_type(8)))  int      i32x8;
typedef __attribute__((ext_vector_type(4)))  int      i32x4;

#define BB 8
#define LL 1024
#define DD 1024
#define HH 16
#define DPH 64
#define MAXREL 32
#define NREL 65

static __device__ __forceinline__ v8f wmma_f16(v16h a, v16h b, v8f c) {
  // D = A(16x32 f16) * B(32x16 f16) + C(16x16 f32)
  return __builtin_amdgcn_wmma_f32_16x16x32_f16(false, a, false, b, (short)0, c, false, false);
}

// ---------------------------------------------------------------------------
// TDM: 2D tile load Global -> LDS (Tensor Data Mover, TENSORcnt-tracked).
// Descriptor bitfields per CDNA5 ISA.
//   tensor: [d1 rows][stride0 elems] row-major, elements of (1<<ds_log2) bytes
//   tile:   [tile_d1 rows][tile_d0 elems] starting at gaddr
//   LDS:    filled sequentially, optional pad of (pad_amt+1) dwords every
//           2^(pad_int+1) dwords.
// ---------------------------------------------------------------------------
static __device__ __forceinline__ void tdm_load_2d(
    unsigned lds_off, const void* gaddr,
    unsigned tensor_d0, unsigned tensor_d1, unsigned long long stride0,
    unsigned tile_d0, unsigned tile_d1, unsigned ds_log2,
    unsigned pad_en, unsigned pad_int, unsigned pad_amt)
{
  unsigned long long ga = (unsigned long long)gaddr;
  u32x4 g0;
  g0[0] = 1u;                                                  // count=1 (user)
  g0[1] = lds_off;                                             // lds_addr
  g0[2] = (unsigned)(ga & 0xFFFFFFFFu);                        // global_addr lo
  g0[3] = (unsigned)((ga >> 32) & 0x01FFFFFFu) | (2u << 30);   // addr hi | type=2
  i32x8 g1;
  g1[0] = (int)(((ds_log2 & 3u) << 16) | ((pad_en & 1u) << 20) |
                ((pad_int & 7u) << 22) | ((pad_amt & 127u) << 25));
  g1[1] = (int)((tensor_d0 & 0xFFFFu) << 16);                  // dim0[15:0]
  g1[2] = (int)(((tensor_d0 >> 16) & 0xFFFFu) | ((tensor_d1 & 0xFFFFu) << 16));
  g1[3] = (int)(((tensor_d1 >> 16) & 0xFFFFu) | ((tile_d0 & 0xFFFFu) << 16));
  g1[4] = (int)(tile_d1 & 0xFFFFu);                            // tile_dim1, dim2=0
  g1[5] = (int)(unsigned)(stride0 & 0xFFFFFFFFull);            // dim0_stride lo
  g1[6] = (int)(unsigned)((stride0 >> 32) & 0xFFFFull);        // dim0_stride hi
  g1[7] = 0;
  i32x4 z4 = {0, 0, 0, 0};
#if defined(__clang_major__) && __clang_major__ >= 23
  i32x8 z8 = {0, 0, 0, 0, 0, 0, 0, 0};
  __builtin_amdgcn_tensor_load_to_lds(g0, g1, z4, z4, z8, 0);
#else
  __builtin_amdgcn_tensor_load_to_lds(g0, g1, z4, z4, 0);
#endif
}

static __device__ __forceinline__ unsigned lds_offset(const void* p) {
  // LDS flat addresses carry the LDS byte offset in addr[31:0]
  return (unsigned)(unsigned long long)p;
}

// ---------------------------------------------------------------------------
// Kernel 1: Y = X @ W^T + bias, scaled, f16 output.
//   vtrans=0: head-major [B,H,L,DPH]   vtrans=1: transposed [B,H,DPH,L]
// One wave computes a 16(M) x 64(N) tile; K loop over D in steps of 32.
// ---------------------------------------------------------------------------
__global__ __launch_bounds__(256) void proj_kernel(
    const float* __restrict__ X, const float* __restrict__ W,
    const float* __restrict__ bias, _Float16* __restrict__ Yh, float scale,
    int vtrans)
{
  const int wave = threadIdx.x >> 5;
  const int lane = threadIdx.x & 31;
  const int gw   = blockIdx.x * 8 + wave;   // 8192 wave tiles
  const int m0   = (gw >> 4) * 16;          // 512 M tiles
  const int n0   = (gw & 15) * 64;          // 16 N groups
  const int lm   = lane & 15;
  const int koff = (lane & 16) ? 16 : 0;    // A-frag K offset
  const int bko  = (lane & 16) ? 16 : 0;    // B-frag K offset
  const int rbase = (lane >> 4) * 8;        // C row base

  v8f c[4] = {v8f{}, v8f{}, v8f{}, v8f{}};
  for (int k0 = 0; k0 < DD; k0 += 32) {
    const float* xrow = X + (size_t)(m0 + lm) * DD + k0 + koff;
    v16h a;
#pragma unroll
    for (int i = 0; i < 16; ++i) a[i] = (_Float16)xrow[i];
#pragma unroll
    for (int t = 0; t < 4; ++t) {
      const float* wrow = W + (size_t)(n0 + t * 16 + lm) * DD + k0 + bko;
      v16h bf;
#pragma unroll
      for (int i = 0; i < 16; ++i) bf[i] = (_Float16)wrow[i];
      c[t] = wmma_f16(a, bf, c[t]);
    }
  }
#pragma unroll
  for (int t = 0; t < 4; ++t) {
    const int n = n0 + t * 16 + lm;
    const float bb = bias[n];
    const int h = n >> 6, d = n & 63;
#pragma unroll
    for (int g = 0; g < 8; ++g) {
      const int mm = m0 + g + rbase;
      const int b  = mm >> 10, q = mm & 1023;
      const float v = (c[t][g] + bb) * scale;
      const size_t idx = vtrans
          ? (((size_t)b * HH + h) * DPH + d) * LL + q    // [B,H,DPH,L]
          : (((size_t)b * HH + h) * LL + q) * DPH + d;   // [B,H,L,DPH]
      Yh[idx] = (_Float16)v;
    }
  }
}

// ---------------------------------------------------------------------------
// Kernel 2: fused attention. One block (8 wave32) per (b,h, 16-row q-strip).
// scores = QK^T + Q.rel -> softmax (attn to d_out)
// ctx    = attn @ V + bucket(attn) @ rel_emb   (V^T chunks staged by TDM,
//          double-buffered with s_wait_tensorcnt)
// ---------------------------------------------------------------------------
__global__ __launch_bounds__(256) void attn_kernel(
    const _Float16* __restrict__ qh, const _Float16* __restrict__ kh,
    const _Float16* __restrict__ vt, const float* __restrict__ rel,
    float* __restrict__ attn_out, _Float16* __restrict__ ctxh)
{
  __shared__ float    relS[NREL][DPH];     // 16.6 KB (TDM-loaded, contiguous)
  __shared__ float    sc[16][1032];        // 66.0 KB score / attn rows
  __shared__ float    qrel[16][80];        //  5.1 KB
  __shared__ float    arel[16][66];        //  4.2 KB
  __shared__ float    ctxrel[16][DPH];     //  4.1 KB
  __shared__ _Float16 VT[2][64][136];      // 34.8 KB V^T chunks [buf][d][k+pad]
  __shared__ float    cpart[4][8][32];     //  4.0 KB k-half partial C

  const int tid  = threadIdx.x;
  const int wave = tid >> 5;
  const int lane = tid & 31;
  const int bh   = blockIdx.x >> 6;           // 0..127
  const int q0   = (blockIdx.x & 63) * 16;    // q strip
  const int lm   = lane & 15;
  const int koff = (lane & 16) ? 16 : 0;
  const int bko  = (lane & 16) ? 16 : 0;
  const int rbase = (lane >> 4) * 8;

  // --- TDM: whole rel_emb table (65x64 f32) -> LDS ---
  if (wave == 0) {
    tdm_load_2d(lds_offset(&relS[0][0]), rel,
                /*tensor*/ DPH, NREL, /*stride0*/ DPH,
                /*tile*/ DPH, NREL, /*4B*/ 2, 0, 0, 0);
    __builtin_amdgcn_s_wait_tensorcnt(0);
  }
  __syncthreads();

  // Q fragments for rows q0..q0+15 (every wave keeps its own copy)
  const _Float16* qrow = qh + ((size_t)bh * LL + q0 + lm) * DPH;
  v16h aq0, aq1;
#pragma unroll
  for (int i = 0; i < 16; ++i) {
    const int ii = i;
    aq0[i] = qrow[koff + ii];
    aq1[i] = qrow[32 + koff + ii];
  }

  // qrel[16][65] = Qtile @ rel^T   (wave 0, 5 N-chunks x 2 WMMAs)
  if (wave == 0) {
    for (int cc = 0; cc < 5; ++cc) {
      const int r = cc * 16 + lm;
      v16h b0, b1;
#pragma unroll
      for (int i = 0; i < 16; ++i) {
        const float v0 = (r < NREL) ? relS[r][bko + i] : 0.f;
        const float v1 = (r < NREL) ? relS[r][32 + bko + i] : 0.f;
        b0[i] = (_Float16)v0;
        b1[i] = (_Float16)v1;
      }
      v8f acc = {};
      acc = wmma_f16(aq0, b0, acc);
      acc = wmma_f16(aq1, b1, acc);
#pragma unroll
      for (int g = 0; g < 8; ++g) qrel[g + rbase][cc * 16 + lm] = acc[g];
    }
  }
  __syncthreads();

  // scores[16][1024] = Qtile @ K^T + qrel[dist]
  for (int cc = wave; cc < 64; cc += 8) {
    const int kp0 = cc * 16;
    const _Float16* krow = kh + ((size_t)bh * LL + kp0 + lm) * DPH;
    if (cc + 8 < 64)
      __builtin_prefetch(kh + ((size_t)bh * LL + (cc + 8) * 16 + lm) * DPH, 0, 1);
    v16h b0, b1;
#pragma unroll
    for (int i = 0; i < 16; ++i) { b0[i] = krow[bko + i]; b1[i] = krow[32 + bko + i]; }
    v8f acc = {};
    acc = wmma_f16(aq0, b0, acc);
    acc = wmma_f16(aq1, b1, acc);
    const int kp = kp0 + lm;
#pragma unroll
    for (int g = 0; g < 8; ++g) {
      const int row  = g + rbase;
      const int qpos = q0 + row;
      int dlt = kp - qpos;
      dlt = (dlt < -MAXREL) ? -MAXREL : (dlt > MAXREL ? MAXREL : dlt);
      sc[row][kp] = acc[g] + qrel[row][dlt + MAXREL];
    }
  }
  __syncthreads();

  // softmax per row (2 rows per wave), write attn to global
  float* attn_base = attn_out + ((size_t)bh * LL + q0) * LL;
#pragma unroll
  for (int rr = 0; rr < 2; ++rr) {
    const int row = wave * 2 + rr;
    float mx = -3.0e38f;
    for (int k = lane; k < LL; k += 32) mx = fmaxf(mx, sc[row][k]);
#pragma unroll
    for (int off = 16; off > 0; off >>= 1) mx = fmaxf(mx, __shfl_xor(mx, off, 32));
    float sum = 0.f;
    for (int k = lane; k < LL; k += 32) {
      const float e = __expf(sc[row][k] - mx);
      sc[row][k] = e;
      sum += e;
    }
#pragma unroll
    for (int off = 16; off > 0; off >>= 1) sum += __shfl_xor(sum, off, 32);
    const float inv = 1.f / sum;
    for (int k = lane; k < LL; k += 32) {
      const float a = sc[row][k] * inv;
      sc[row][k] = a;
      attn_base[(size_t)row * LL + k] = a;
    }
  }
  __syncthreads();

  // bucket-aggregate attn by relative distance, then ctx_rel = arel @ rel_emb
  for (int i = tid; i < 16 * 66; i += 256) arel[i / 66][i % 66] = 0.f;
  __syncthreads();
#pragma unroll
  for (int rr = 0; rr < 2; ++rr) {
    const int row  = wave * 2 + rr;
    const int qpos = q0 + row;
    for (int k = lane; k < LL; k += 32) {
      int dlt = k - qpos;
      dlt = (dlt < -MAXREL) ? -MAXREL : (dlt > MAXREL ? MAXREL : dlt);
      atomicAdd(&arel[row][dlt + MAXREL], sc[row][k]);
    }
  }
  __syncthreads();
#pragma unroll
  for (int rr = 0; rr < 2; ++rr) {
    const int row = wave * 2 + rr;
#pragma unroll
    for (int dd = 0; dd < 2; ++dd) {
      const int d = lane + dd * 32;
      float acc = 0.f;
      for (int r = 0; r < NREL; ++r) acc += arel[row][r] * relS[r][d];
      ctxrel[row][d] = acc;
    }
  }
  __syncthreads();

  // ctx[16][64] = attn @ V.
  // wave = nchunk(0..3) + 4*khalf ; V^T chunks [64 d][128 k] via TDM, dbl-buffered.
  const int nchunk = wave & 3;
  const int khalf  = wave >> 2;
  const _Float16* vbase = vt + (size_t)bh * DPH * LL;   // [64 rows][1024] f16

  if (wave == 0) {
    // prologue: chunk 0 -> buffer 0 (pad 4 dwords every 64 dwords -> stride 136)
    tdm_load_2d(lds_offset(&VT[0][0][0]), vbase,
                /*tensor*/ LL, DPH, /*stride0*/ LL,
                /*tile*/ 128, DPH, /*2B*/ 1, 1, 5, 3);
  }
  v8f cacc = {};
  for (int it = 0; it < 8; ++it) {
    if (wave == 0) {
      if (it < 7) {
        const _Float16* gsrc = vbase + (it + 1) * 128;
        tdm_load_2d(lds_offset(&VT[(it + 1) & 1][0][0]), gsrc,
                    LL, DPH, LL, 128, DPH, 1, 1, 5, 3);
        __builtin_amdgcn_s_wait_tensorcnt(1);   // chunk `it` complete
      } else {
        __builtin_amdgcn_s_wait_tensorcnt(0);
      }
    }
    __syncthreads();
    const int buf = it & 1;
    const int kb  = it * 128;
#pragma unroll
    for (int s = 0; s < 2; ++s) {
      const int kk = khalf * 64 + s * 32;       // within 128-chunk
      v16h a, b;
#pragma unroll
      for (int i = 0; i < 16; ++i) {
        const int ii = i;
        a[i] = (_Float16)sc[lm][kb + kk + koff + ii];
        b[i] = VT[buf][nchunk * 16 + lm][kk + bko + i];
      }
      cacc = wmma_f16(a, b, cacc);
    }
    __syncthreads();
  }

  // combine k-halves: waves 4..7 publish partials, waves 0..3 reduce + store
  if (khalf == 1) {
#pragma unroll
    for (int g = 0; g < 8; ++g) cpart[nchunk][g][lane] = cacc[g];
  }
  __syncthreads();
  if (khalf == 0) {
#pragma unroll
    for (int g = 0; g < 8; ++g) {
      const int row = g + rbase;
      const int d   = nchunk * 16 + lm;
      const float v = cacc[g] + cpart[nchunk][g][lane] + ctxrel[row][d];
      ctxh[((size_t)bh * LL + q0 + row) * DPH + d] = (_Float16)v;
    }
  }
}

// ---------------------------------------------------------------------------
// Kernel 3: out = ctx @ Wo^T + bo  (ctx is f16 head-major, out f32 [B,L,D])
// ---------------------------------------------------------------------------
__global__ __launch_bounds__(256) void out_proj_kernel(
    const _Float16* __restrict__ ctxh, const float* __restrict__ W,
    const float* __restrict__ bias, float* __restrict__ Y)
{
  const int wave = threadIdx.x >> 5;
  const int lane = threadIdx.x & 31;
  const int gw   = blockIdx.x * 8 + wave;
  const int m0   = (gw >> 4) * 16;
  const int n0   = (gw & 15) * 64;
  const int lm   = lane & 15;
  const int koff = (lane & 16) ? 16 : 0;
  const int bko  = (lane & 16) ? 16 : 0;
  const int rbase = (lane >> 4) * 8;
  const int m = m0 + lm;
  const int bI = m >> 10, q = m & 1023;

  v8f c[4] = {v8f{}, v8f{}, v8f{}, v8f{}};
  for (int k0 = 0; k0 < DD; k0 += 32) {
    const int h = k0 >> 6, db = k0 & 63;
    const _Float16* arow = ctxh + (((size_t)bI * HH + h) * LL + q) * DPH + db + koff;
    v16h a;
#pragma unroll
    for (int i = 0; i < 16; ++i) a[i] = arow[i];
#pragma unroll
    for (int t = 0; t < 4; ++t) {
      const float* wrow = W + (size_t)(n0 + t * 16 + lm) * DD + k0 + bko;
      v16h bf;
#pragma unroll
      for (int i = 0; i < 16; ++i) bf[i] = (_Float16)wrow[i];
      c[t] = wmma_f16(a, bf, c[t]);
    }
  }
#pragma unroll
  for (int t = 0; t < 4; ++t) {
    const int n = n0 + t * 16 + lm;
    const float bb = bias[n];
#pragma unroll
    for (int g = 0; g < 8; ++g) {
      const int mm = m0 + g + rbase;
      Y[(size_t)mm * DD + n] = c[t][g] + bb;
    }
  }
}

// ---------------------------------------------------------------------------
extern "C" void kernel_launch(void* const* d_in, const int* in_sizes, int n_in,
                              void* d_out, int out_size, void* d_ws, size_t ws_size,
                              hipStream_t stream) {
  const float* key   = (const float*)d_in[0];
  const float* value = (const float*)d_in[1];
  const float* query = (const float*)d_in[2];
  const float* Wq = (const float*)d_in[3];
  const float* bq = (const float*)d_in[4];
  const float* Wk = (const float*)d_in[5];
  const float* bk = (const float*)d_in[6];
  const float* Wv = (const float*)d_in[7];
  const float* bv = (const float*)d_in[8];
  const float* Wo = (const float*)d_in[9];
  const float* bo = (const float*)d_in[10];
  const float* rel = (const float*)d_in[11];

  float* out  = (float*)d_out;                      // B*L*D f32
  float* attn = out + (size_t)BB * LL * DD;         // B*H*L*L f32

  const size_t headElems = (size_t)BB * HH * LL * DPH;
  _Float16* qhp  = (_Float16*)d_ws;
  _Float16* khp  = qhp + headElems;
  _Float16* vtp  = khp + headElems;                 // V^T head-major [B,H,DPH,L]
  _Float16* ctxh = vtp + headElems;

  dim3 blk(256);
  dim3 gProj(1024);                                 // 8192 wave tiles / 8
  proj_kernel<<<gProj, blk, 0, stream>>>(query, Wq, bq, qhp, 0.125f, 0);
  proj_kernel<<<gProj, blk, 0, stream>>>(key,   Wk, bk, khp, 1.0f,   0);
  proj_kernel<<<gProj, blk, 0, stream>>>(value, Wv, bv, vtp, 1.0f,   1);

  attn_kernel<<<dim3(BB * HH * (LL / 16)), blk, 0, stream>>>(qhp, khp, vtp, rel, attn, ctxh);

  out_proj_kernel<<<gProj, blk, 0, stream>>>(ctxh, Wo, bo, out);
}
